// DCRNN_38577396252967
// MI455X (gfx1250) — matrix operand order repaired and checked by reading, not compiled, not run on hardware
//
#include <hip/hip_runtime.h>
#include <hip/hip_bf16.h>

typedef __attribute__((ext_vector_type(16))) __bf16 v16bf;
typedef __attribute__((ext_vector_type(8)))  __bf16 v8bf;
typedef __attribute__((ext_vector_type(8)))  float  v8f;

#define TILE_M 128
#define TILE_N 64
#define TILE_K 32
#define LDS_PITCH 40   // 80 B/row: 16B-aligned rows, bank-conflict-free frag reads
#define FP      128    // padded feature pitch for all activation buffers

// ---- CDNA5 async global->LDS staging (ASYNCcnt-tracked; ISA 10.7 / 15.18.3) ----
__device__ __forceinline__ void async_lds_b128(unsigned lds_addr, unsigned long long gaddr) {
    asm volatile("global_load_async_to_lds_b128 %0, %1, off"
                 :: "v"(lds_addr), "v"(gaddr) : "memory");
}
__device__ __forceinline__ void wait_async0() {
    asm volatile("s_wait_asynccnt 0x0" ::: "memory");
}

// C[M,Nc](f32) = A_bf16[M,K] x B_bf16[K,Nc], all dims tile-aligned (no guards):
// M%128==0, Nc%64==0, K%32==0, 16B-aligned rows. beta=1 accumulates; optional
// bias + activation (1=sigmoid, 2=tanh). grid=(Nc/64, M/128, batch).
__global__ __launch_bounds__(256) void dcrnn_gemm_bf16(
    const __bf16* __restrict__ A, long lda, long sA,
    const __bf16* __restrict__ Bm, long ldb, long sB,
    float* __restrict__ C, long ldc, long sC,
    int K, int beta, const float* __restrict__ bias, int act)
{
    __shared__ __bf16 As[2][TILE_M * LDS_PITCH];
    __shared__ __bf16 Bs[2][TILE_N * LDS_PITCH];

    const int tid  = threadIdx.x;
    const int lane = tid & 31;
    const int wave = tid >> 5;     // 0..7
    const int wm   = wave >> 1;    // 0..3 (M, 4 x 32)
    const int wn   = wave & 1;     // 0..1 (N, 2 x 32)
    const int m0   = blockIdx.y * TILE_M;
    const int n0   = blockIdx.x * TILE_N;

    const __bf16* Ab = A  + (size_t)blockIdx.z * sA;
    const __bf16* Bb = Bm + (size_t)blockIdx.z * sB;
    float*        Cb = C  + (size_t)blockIdx.z * sC;

    v8f acc[2][2];
#pragma unroll
    for (int i = 0; i < 2; ++i)
#pragma unroll
        for (int j = 0; j < 2; ++j)
#pragma unroll
            for (int r = 0; r < 8; ++r) acc[i][j][r] = 0.0f;

    // A staging: 2 threads/row, 16 bf16 (32B) each -> 2 async b128 per thread
    const int a_row = tid >> 1;
    const int a_k   = (tid & 1) * 16;
    const __bf16* aptr = Ab + (size_t)(m0 + a_row) * lda + a_k;
    const unsigned ldsA[2] = {
        (unsigned)(size_t)&As[0][a_row * LDS_PITCH + a_k],
        (unsigned)(size_t)&As[1][a_row * LDS_PITCH + a_k] };
    auto issueA = [&](int k0, int buf) {
        const unsigned long long g = (unsigned long long)(size_t)(aptr + k0);
        async_lds_b128(ldsA[buf],      g);
        async_lds_b128(ldsA[buf] + 16, g + 16);
    };

    // B staging: 8 threads/row(k), 8 cols (16B) each; transposed into LDS[n][k]
    const int b_k = tid >> 3;
    const int b_n = (tid & 7) * 8;
    const __bf16* bptr = Bb + (size_t)b_k * ldb + n0 + b_n;
    unsigned breg[4];
    auto fetchB = [&](int k0) {
        const unsigned* s4 = (const unsigned*)(bptr + (size_t)k0 * ldb);
#pragma unroll
        for (int j = 0; j < 4; ++j) breg[j] = s4[j];
    };
    auto storeB = [&](int buf) {
        const __bf16* t = (const __bf16*)breg;
#pragma unroll
        for (int j = 0; j < 8; ++j) Bs[buf][(b_n + j) * LDS_PITCH + b_k] = t[j];
    };

    // WMMA fragment addressing (ISA 7.12.2, bf16)
    const int lrow   = lane & 15;
    const int kbaseA = (lane & 16) ? 8 : 0;
    const int kbaseB = (lane & 16) ? 8 : 0;
    const int mhi    = (lane >> 4) * 8;

    // ---- software-pipelined, double-buffered, one barrier per K-step ----
    issueA(0, 0);
    fetchB(0);
    storeB(0);

    int buf = 0;
    for (int k0 = 0; k0 < K; k0 += TILE_K) {
        wait_async0();        // this wave's async writes into As[buf] done
        __syncthreads();      // everyone's As/Bs[buf] stores visible
        if (k0 + TILE_K < K) {
            issueA(k0 + TILE_K, buf ^ 1);
            fetchB(k0 + TILE_K);
            storeB(buf ^ 1);
        }

        v16bf afrag[2], bfrag[2];
#pragma unroll
        for (int fm = 0; fm < 2; ++fm) {
            const __bf16* ap = &As[buf][(wm * 32 + fm * 16 + lrow) * LDS_PITCH];
            v8bf lo = *(const v8bf*)(ap + kbaseA);        // K = kbaseA .. +7
            v8bf hi = *(const v8bf*)(ap + 16 + kbaseA);   // K = 16+kbaseA .. +7
            afrag[fm] = __builtin_shufflevector(lo, hi,
                0,1,2,3,4,5,6,7,8,9,10,11,12,13,14,15);
        }
#pragma unroll
        for (int fn = 0; fn < 2; ++fn) {
            const __bf16* bp = &Bs[buf][(wn * 32 + fn * 16 + lrow) * LDS_PITCH + kbaseB];
            v8bf lo = *(const v8bf*)(bp);
            v8bf hi = *(const v8bf*)(bp + 16);
            bfrag[fn] = __builtin_shufflevector(lo, hi,
                0,1,2,3,4,5,6,7,8,9,10,11,12,13,14,15);
        }
#pragma unroll
        for (int fm = 0; fm < 2; ++fm)
#pragma unroll
            for (int fn = 0; fn < 2; ++fn)
                acc[fm][fn] = __builtin_amdgcn_wmma_f32_16x16x32_bf16(
                    false, afrag[fm], false, bfrag[fn],
                    (short)0, acc[fm][fn], false, false);
        buf ^= 1;
    }

    // ---- epilogue (tile-aligned: no guards) ----
#pragma unroll
    for (int fm = 0; fm < 2; ++fm) {
#pragma unroll
        for (int fn = 0; fn < 2; ++fn) {
            const int gn = n0 + wn * 32 + fn * 16 + lrow;
            const int mb = m0 + wm * 32 + fm * 16 + mhi;
#pragma unroll
            for (int r = 0; r < 8; ++r) {
                float v = acc[fm][fn][r];
                const size_t idx = (size_t)(mb + r) * ldc + gn;
                if (beta) v += Cb[idx];
                if (bias) v += bias[gn];
                if (act == 1)      v = 1.0f / (1.0f + __expf(-v));
                else if (act == 2) v = tanhf(v);
                Cb[idx] = v;
            }
        }
    }
}

// 4-wide f32 -> bf16 (n % 4 == 0, 16B-aligned buffers)
__global__ void dcrnn_cvt4(const float* __restrict__ in, __bf16* __restrict__ out, long n4)
{
    const long i = (long)blockIdx.x * 256 + threadIdx.x;
    if (i >= n4) return;
    const float4 v = ((const float4*)in)[i];
    __bf16 o[4] = { (__bf16)v.x, (__bf16)v.y, (__bf16)v.z, (__bf16)v.w };
    ((uint2*)out)[i] = *(const uint2*)o;
}

// Pack W[(nm*f_real), od] -> bf16 [nm][FP][od], zero rows f_real..FP-1 per block
__global__ void dcrnn_pack_w(const float* __restrict__ W, __bf16* __restrict__ out,
                             int f_real, int od, long total)
{
    const long i = (long)blockIdx.x * 256 + threadIdx.x;
    if (i >= total) return;
    const int  c   = (int)(i % od);
    const long ro  = i / od;
    const int  r   = (int)(ro % FP);
    const long blk = ro / FP;
    out[i] = (r < f_real) ? (__bf16)W[(blk * f_real + r) * od + c] : (__bf16)0.0f;
}

// xh[b,n,j] = j<F ? xsrc(b,n,j) : j<F+H ? h[b,n,j-F] : 0   (bf16, pitch FP)
__global__ void dcrnn_build_xh(const float* __restrict__ xsrc, long sx_b, long sx_n, long xoff,
                               const float* __restrict__ h, int F, int H,
                               __bf16* __restrict__ out, int N, long total)
{
    const long i = (long)blockIdx.x * 256 + threadIdx.x;
    if (i >= total) return;
    const int  j  = (int)(i & (FP - 1));
    const long bn = i >> 7;
    float v = 0.0f;
    if (j < F) {
        const long n = bn % N, b = bn / N;
        v = xsrc[b * sx_b + n * sx_n + xoff + j];
    } else if (j < F + H) {
        v = h[bn * H + (j - F)];
    }
    out[i] = (__bf16)v;
}

// xc pad-feature: j<F ? xsrc : j<F+H ? r*h : 0   (r = zr[...,0:H], post-sigmoid)
__global__ void dcrnn_build_xc(const float* __restrict__ xsrc, long sx_b, long sx_n, long xoff,
                               const float* __restrict__ h, const float* __restrict__ zr,
                               int F, int H, __bf16* __restrict__ out, int N, long total)
{
    const long i = (long)blockIdx.x * 256 + threadIdx.x;
    if (i >= total) return;
    const int  j  = (int)(i & (FP - 1));
    const long bn = i >> 7;
    float v = 0.0f;
    if (j < F) {
        const long n = bn % N, b = bn / N;
        v = xsrc[b * sx_b + n * sx_n + xoff + j];
    } else if (j < F + H) {
        const int k = j - F;
        v = zr[bn * 2 * H + k] * h[bn * H + k];
    }
    out[i] = (__bf16)v;
}

// h = u*h + (1-u)*c ; u = zr[..., H:2H]. Optionally scatter into out[B,T,N,H].
__global__ void dcrnn_update_h(const float* __restrict__ zr, const float* __restrict__ c,
                               float* __restrict__ h, float* __restrict__ out_seq,
                               long t, long T, int H, int N, long total)
{
    const long i = (long)blockIdx.x * 256 + threadIdx.x;
    if (i >= total) return;
    const long bn = i / H;
    const int  k  = (int)(i % H);
    const float u  = zr[bn * 2 * H + H + k];
    const float hn = u * h[i] + (1.0f - u) * c[i];
    h[i] = hn;
    if (out_seq) {
        const long n = bn % N, b = bn / N;
        out_seq[((b * T + t) * N + n) * (long)H + k] = hn;
    }
}

extern "C" void kernel_launch(void* const* d_in, const int* in_sizes, int n_in,
                              void* d_out, int out_size, void* d_ws, size_t ws_size,
                              hipStream_t stream)
{
    (void)in_sizes; (void)n_in; (void)out_size;
    constexpr int  Bsz = 64, T = 24, N = 1024, F = 2, H = 64, S = 2;
    constexpr long BN  = (long)Bsz * N;   // 65536
    constexpr int  f0  = F + H;           // 66 (padded to FP=128)
    constexpr int  f1  = H + H;           // 128
    constexpr int  NM  = 2 * S + 1;       // 5 diffusion feature blocks

    const float* x    = (const float*)d_in[0];
    const float* init = (const float*)d_in[1];
    const float* sup  = (const float*)d_in[2];
    const float* Wg0  = (const float*)d_in[3]; const float* bg0 = (const float*)d_in[4];
    const float* Wc0  = (const float*)d_in[5]; const float* bc0 = (const float*)d_in[6];
    const float* Wg1  = (const float*)d_in[7]; const float* bg1 = (const float*)d_in[8];
    const float* Wc1  = (const float*)d_in[9]; const float* bc1 = (const float*)d_in[10];
    float* out = (float*)d_out;

    // ---- carve workspace (all 256B-aligned) ----
    char* p = (char*)d_ws;
    auto carve = [&](size_t bytes) -> void* {
        void* r = (void*)p; p += (bytes + 255) & ~(size_t)255; return r;
    };
    __bf16* sup_bf = (__bf16*)carve((size_t)S * N * N * 2);
    __bf16* wg0_bf = (__bf16*)carve((size_t)NM * FP * 2 * H * 2);
    __bf16* wc0_bf = (__bf16*)carve((size_t)NM * FP * H * 2);
    __bf16* wg1_bf = (__bf16*)carve((size_t)NM * FP * 2 * H * 2);
    __bf16* wc1_bf = (__bf16*)carve((size_t)NM * FP * H * 2);
    float*  h0     = (float*) carve((size_t)BN * H * 4);
    float*  h1     = (float*) carve((size_t)BN * H * 4);
    __bf16* xh_bf  = (__bf16*)carve((size_t)BN * FP * 2);
    __bf16* bfA    = (__bf16*)carve((size_t)BN * FP * 2);
    __bf16* bfB    = (__bf16*)carve((size_t)BN * FP * 2);
    float*  tmpf   = (float*) carve((size_t)BN * FP * 4);
    float*  zr     = (float*) carve((size_t)BN * 2 * H * 4);
    float*  cbuf   = (float*) carve((size_t)BN * H * 4);
    if ((size_t)(p - (char*)d_ws) > ws_size) return;

    auto cvt = [&](const float* src, __bf16* dst, long n) {
        const long n4 = n / 4;
        dcrnn_cvt4<<<(unsigned)((n4 + 255) / 256), 256, 0, stream>>>(src, dst, n4);
    };
    auto packw = [&](const float* W, __bf16* dst, int f_real, int od) {
        const long total = (long)NM * FP * od;
        dcrnn_pack_w<<<(unsigned)((total + 255) / 256), 256, 0, stream>>>(W, dst, f_real, od, total);
    };
    auto gemm = [&](const __bf16* A, long lda, long sA,
                    const __bf16* Bm, long ldb, long sB,
                    float* C, long ldc, long sC,
                    int M, int Nc, int K, int batch,
                    int beta, const float* bias, int act) {
        dim3 g((unsigned)(Nc / TILE_N), (unsigned)(M / TILE_M), (unsigned)batch);
        dcrnn_gemm_bf16<<<g, 256, 0, stream>>>(A, lda, sA, Bm, ldb, sB,
                                               C, ldc, sC, K, beta, bias, act);
    };

    // gconv: outbuf[BN, od] = concat(f, A0 f, A0^2 f, A1 f, A1^2 f) @ Wpad + b, then act
    auto gconv = [&](float* outbuf, const __bf16* feat0, int od,
                     const __bf16* Wp, const float* bias, int act) {
        gemm(feat0, FP, 0, Wp, od, 0, outbuf, od, 0,
             (int)BN, od, FP, 1, /*beta=*/0, nullptr, 0);
        int idx = 1;
        for (int s = 0; s < S; ++s) {
            const __bf16* cur = feat0;
            for (int k = 0; k < 2; ++k) {
                __bf16* nxt = (cur == bfA) ? bfB : bfA;
                // diffusion: tmpf[b] = support_s (NxN) x cur[b] (N x FP)
                gemm(sup_bf + (size_t)s * N * N, N, 0,
                     cur, FP, (long)N * FP,
                     tmpf, FP, (long)N * FP,
                     N, FP, N, Bsz, 0, nullptr, 0);
                cvt(tmpf, nxt, BN * (long)FP);
                const int last = (idx == NM - 1);
                gemm(nxt, FP, 0, Wp + (size_t)idx * FP * od, od, 0,
                     outbuf, od, 0, (int)BN, od, FP, 1,
                     /*beta=*/1, last ? bias : nullptr, last ? act : 0);
                cur = nxt; ++idx;
            }
        }
    };

    // ---- once-per-launch conversions & state init ----
    cvt(sup, sup_bf, (long)S * N * N);
    packw(Wg0, wg0_bf, f0, 2 * H);
    packw(Wc0, wc0_bf, f0, H);
    packw(Wg1, wg1_bf, f1, 2 * H);
    packw(Wc1, wc1_bf, f1, H);
    hipMemcpyAsync(h0, init,                  (size_t)BN * H * 4, hipMemcpyDeviceToDevice, stream);
    hipMemcpyAsync(h1, init + (size_t)BN * H, (size_t)BN * H * 4, hipMemcpyDeviceToDevice, stream);

    const long elemsH = BN * (long)H;
    const long elemsP = BN * (long)FP;
    const unsigned gP = (unsigned)((elemsP + 255) / 256);
    const unsigned gH = (unsigned)((elemsH + 255) / 256);

    // ---- fused two-layer time loop ----
    for (int t = 0; t < T; ++t) {
        // ===== layer 0: input x[:, t] over [B,T,N,F] =====
        const long sx_b0 = (long)T * N * F, sx_n0 = F, xoff0 = (long)t * N * F;
        dcrnn_build_xh<<<gP, 256, 0, stream>>>(x, sx_b0, sx_n0, xoff0, h0, F, H,
                                               xh_bf, N, elemsP);
        gconv(zr, xh_bf, 2 * H, wg0_bf, bg0, /*sigmoid*/1);
        dcrnn_build_xc<<<gP, 256, 0, stream>>>(x, sx_b0, sx_n0, xoff0, h0, zr, F, H,
                                               xh_bf, N, elemsP);
        gconv(cbuf, xh_bf, H, wc0_bf, bc0, /*tanh*/2);
        dcrnn_update_h<<<gH, 256, 0, stream>>>(zr, cbuf, h0, nullptr, t, T, H, N, elemsH);

        // ===== layer 1: input = h0 (just updated) =====
        const long sx_b1 = (long)N * H, sx_n1 = H;
        dcrnn_build_xh<<<gP, 256, 0, stream>>>(h0, sx_b1, sx_n1, 0, h1, H, H,
                                               xh_bf, N, elemsP);
        gconv(zr, xh_bf, 2 * H, wg1_bf, bg1, 1);
        dcrnn_build_xc<<<gP, 256, 0, stream>>>(h0, sx_b1, sx_n1, 0, h1, zr, H, H,
                                               xh_bf, N, elemsP);
        gconv(cbuf, xh_bf, H, wc1_bf, bc1, 2);
        dcrnn_update_h<<<gH, 256, 0, stream>>>(zr, cbuf, h1, out, t, T, H, N, elemsH);
    }

    // ---- trailing stacked last states: out[B*T*N*H ..] = [h0, h1] ----
    float* lasts = out + (size_t)Bsz * T * N * H;
    hipMemcpyAsync(lasts,                  h0, (size_t)BN * H * 4, hipMemcpyDeviceToDevice, stream);
    hipMemcpyAsync(lasts + (size_t)BN * H, h1, (size_t)BN * H * 4, hipMemcpyDeviceToDevice, stream);
}
